// MultiHeadSelfAttention_75350906241810
// MI455X (gfx1250) — matrix operand order repaired and checked by reading, not compiled, not run on hardware
//
#include <hip/hip_runtime.h>
#include <hip/hip_bf16.h>

#define DEVINL __device__ __forceinline__

typedef __attribute__((ext_vector_type(16))) __bf16 bf16x16;
typedef __attribute__((ext_vector_type(8)))  __bf16 bf16x8;
typedef __attribute__((ext_vector_type(8)))  float  f32x8;
typedef __attribute__((ext_vector_type(4)))  unsigned int u32x4;
typedef __attribute__((ext_vector_type(4)))  int i32x4;
typedef __attribute__((ext_vector_type(8)))  int i32x8;

static constexpr int BATCH = 2;
static constexpr int SEQ   = 4096;
static constexpr int HID   = 1024;
static constexpr int NH    = 16;
static constexpr int HD    = 64;   // head dim

// ---- Tensor Data Mover availability (device pass only; host pass -> 0) ----
#if defined(__gfx1250__) && __has_builtin(__builtin_amdgcn_tensor_load_to_lds) && \
    __has_builtin(__builtin_amdgcn_s_wait_tensorcnt)
#define USE_TDM 1
#if __has_include(<hip/amd_detail/amd_gfx1250_TDM.h>)
#define TDM_SIX_ARGS 1   // therock headers -> clang-23 6-arg builtin
#else
#define TDM_SIX_ARGS 0   // ROCm 7.2 clang-22 5-arg builtin
#endif
#else
#define USE_TDM 0
#endif

union frag16 { bf16x16 v; bf16x8 h[2]; };

// Compose a 16-element bf16 fragment from two contiguous 8-element (16B) runs.
DEVINL bf16x16 ld_frag(const __bf16* p0, const __bf16* p1) {
  frag16 u;
  u.h[0] = *reinterpret_cast<const bf16x8*>(p0);
  u.h[1] = *reinterpret_cast<const bf16x8*>(p1);
  return u.v;
}

DEVINL f32x8 wmma_bf16(bf16x16 a, bf16x16 b, f32x8 c) {
  // D = A(16x32) * B(32x16) + C, f32 accumulate
  return __builtin_amdgcn_wmma_f32_16x16x32_bf16(
      /*neg_a=*/false, a, /*neg_b=*/false, b,
      /*c_mod=*/(short)0, c, /*reuse_a=*/false, /*reuse_b=*/false);
}

#if USE_TDM
// Issue a 2D TDM tile load (bf16 elements): global (tile_h rows x tile_w
// elems, row stride stride_elems) -> LDS (packed row-major, tile_w*2 B/row).
// D# bit layout per CDNA5 ISA ch.8 (group0 128b, group1 256b, groups2/3 = 0).
DEVINL void tdm_load_2d(unsigned lds_addr, const void* gaddr,
                        unsigned tile_w, unsigned tile_h,
                        unsigned stride_elems) {
  const unsigned long long ga = (unsigned long long)(uintptr_t)gaddr;
  u32x4 g0;
  g0[0] = 1u;                                  // count=1, user descriptor
  g0[1] = lds_addr;                            // lds_addr (bytes)
  g0[2] = (unsigned)ga;                        // global_addr[31:0]
  g0[3] = (unsigned)((ga >> 32) & 0x01FFFFFFu) // global_addr[56:32]
          | (2u << 30);                        // type = 2 ("image")
  i32x8 g1;
  g1[0] = (int)(1u << 16);                     // data_size = 1 -> 2 bytes
  g1[1] = (int)((tile_w & 0xFFFFu) << 16);     // tensor_dim0[15:0]
  g1[2] = (int)((tile_w >> 16) | ((tile_h & 0xFFFFu) << 16)); // td0 hi|td1 lo
  g1[3] = (int)((tile_h >> 16) | ((tile_w & 0xFFFFu) << 16)); // td1 hi|tile_dim0
  g1[4] = (int)(tile_h & 0xFFFFu);             // tile_dim1 (tile_dim2 = 0)
  g1[5] = (int)stride_elems;                   // tensor_dim0_stride[31:0]
  g1[6] = 0;                                   // stride hi | dim1_stride lo
  g1[7] = 0;
  i32x4 z4 = {};
#if TDM_SIX_ARGS
  i32x8 z8 = {};
  __builtin_amdgcn_tensor_load_to_lds(g0, g1, z4, z4, z8, 0);
#else
  __builtin_amdgcn_tensor_load_to_lds(g0, g1, z4, z4, 0);
#endif
}
#endif  // USE_TDM

// ---------------------------------------------------------------------------
// f32 -> bf16 conversion (grid-stride)
// ---------------------------------------------------------------------------
__global__ __launch_bounds__(256) void cvt_f32_bf16_kernel(
    const float* __restrict__ src, __bf16* __restrict__ dst, int n) {
  int i = blockIdx.x * blockDim.x + threadIdx.x;
  const int stride = gridDim.x * blockDim.x;
  for (; i < n; i += stride) dst[i] = (__bf16)src[i];
}

// ---------------------------------------------------------------------------
// C[m][n] = sum_k A[m][k] * W[n][k] + bias[n]      (i.e. A @ W^T + b)
// A: [M,K] bf16 row-major, W: [N,K] bf16 row-major.
// OUT_MODE 0: bf16 [M,N] row-major
// OUT_MODE 1: f32  [M,N] row-major
// OUT_MODE 2: bf16 per-head transposed V layout [B, NH, HD, SEQ]
// Block = 128 threads = 4 waves; block tile 64x64, wave tile 32x32 (2x2).
// K-loop is software-pipelined one stage deep so s_wait_loadcnt only covers
// the previous stage's loads while the current WMMAs issue.
// ---------------------------------------------------------------------------
template <int OUT_MODE>
__global__ __launch_bounds__(128) void gemm_xwt_kernel(
    const __bf16* __restrict__ A, const __bf16* __restrict__ W,
    const float* __restrict__ bias, void* __restrict__ out,
    int M, int N, int K) {
  const int lane = threadIdx.x & 31;
  const int wid  = threadIdx.x >> 5;
  const int half = lane >> 4;
  const int lm   = lane & 15;

  const int m_base = blockIdx.x * 64 + (wid >> 1) * 32;
  const int n_base = blockIdx.y * 64 + (wid & 1) * 32;

  f32x8 c[2][2] = {};

  const int kbA = half * 8;   // A-frag run start within 32-wide K chunk
  const int kbB = half * 8;   // B-frag run start within 32-wide K chunk

  const __bf16* arow0 = A + (size_t)(m_base + lm) * K;
  const __bf16* arow1 = A + (size_t)(m_base + 16 + lm) * K;
  const __bf16* wrow0 = W + (size_t)(n_base + lm) * K;
  const __bf16* wrow1 = W + (size_t)(n_base + 16 + lm) * K;

  bf16x16 a0 = ld_frag(arow0 + kbA, arow0 + kbA + 16);
  bf16x16 a1 = ld_frag(arow1 + kbA, arow1 + kbA + 16);
  bf16x16 b0 = ld_frag(wrow0 + kbB, wrow0 + kbB + 16);
  bf16x16 b1 = ld_frag(wrow1 + kbB, wrow1 + kbB + 16);

  for (int k0 = 32; k0 < K; k0 += 32) {
    bf16x16 na0 = ld_frag(arow0 + k0 + kbA, arow0 + k0 + kbA + 16);
    bf16x16 na1 = ld_frag(arow1 + k0 + kbA, arow1 + k0 + kbA + 16);
    bf16x16 nb0 = ld_frag(wrow0 + k0 + kbB, wrow0 + k0 + kbB + 16);
    bf16x16 nb1 = ld_frag(wrow1 + k0 + kbB, wrow1 + k0 + kbB + 16);
    c[0][0] = wmma_bf16(a0, b0, c[0][0]);
    c[0][1] = wmma_bf16(a0, b1, c[0][1]);
    c[1][0] = wmma_bf16(a1, b0, c[1][0]);
    c[1][1] = wmma_bf16(a1, b1, c[1][1]);
    a0 = na0; a1 = na1; b0 = nb0; b1 = nb1;
  }
  c[0][0] = wmma_bf16(a0, b0, c[0][0]);
  c[0][1] = wmma_bf16(a0, b1, c[0][1]);
  c[1][0] = wmma_bf16(a1, b0, c[1][0]);
  c[1][1] = wmma_bf16(a1, b1, c[1][1]);

  for (int ni = 0; ni < 2; ++ni) {
    const int col = n_base + ni * 16 + lm;
    const float bv = bias[col];
    for (int mi = 0; mi < 2; ++mi) {
      for (int r = 0; r < 8; ++r) {
        const int m = m_base + mi * 16 + half * 8 + r;
        const float v = c[mi][ni][r] + bv;
        if (OUT_MODE == 0) {
          ((__bf16*)out)[(size_t)m * N + col] = (__bf16)v;
        } else if (OUT_MODE == 1) {
          ((float*)out)[(size_t)m * N + col] = v;
        } else {
          const int bidx = m >> 12;   // m / SEQ
          const int s    = m & (SEQ - 1);
          const int hh   = col >> 6;  // col / HD
          const int d    = col & (HD - 1);
          ((__bf16*)out)[(((size_t)bidx * NH + hh) * HD + d) * SEQ + s] =
              (__bf16)v;
        }
      }
    }
  }
}

// ---------------------------------------------------------------------------
// Flash attention. One wave (32 threads) per (batch, head, 16-query tile).
// Q,K: [B*S, H] bf16 row-major.  V: [B, NH, HD, S] bf16 (per-head transposed).
// mask: [B,1,S,S] int32.  O: [B*S, H] bf16.
// TDM path: K/V tiles are double-buffered in LDS via tensor_load_to_lds;
// s_wait_tensorcnt(2) overlaps the next pair with the current compute.
// ---------------------------------------------------------------------------
__global__ __launch_bounds__(32) void flash_attn_kernel(
    const __bf16* __restrict__ Q, const __bf16* __restrict__ Kmat,
    const __bf16* __restrict__ Vt, const int* __restrict__ mask,
    __bf16* __restrict__ O) {
  const int lane = threadIdx.x & 31;
  const int half = lane >> 4;
  const int lm   = lane & 15;

  const int q0 = blockIdx.x * 16;
  const int h  = blockIdx.y;
  const int b  = blockIdx.z;

  __shared__ __align__(16) __bf16 pbuf[16][32];
#if USE_TDM
  __shared__ __align__(16) __bf16 ktile[2][32][64];  // [buf][kv][d]
  __shared__ __align__(16) __bf16 vtile[2][64][32];  // [buf][d][kv]
#endif

  const int kbA = half * 8;
  const int kbB = half * 8;

  // Q fragments: 16x64 = two 16x32 K-chunks (A layout)
  const __bf16* qrow = Q + ((size_t)(b * SEQ + q0 + lm) * HID + h * HD);
  const bf16x16 qa0 = ld_frag(qrow + kbA,      qrow + kbA + 16);
  const bf16x16 qa1 = ld_frag(qrow + 32 + kbA, qrow + 32 + kbA + 16);

  f32x8 oacc[4] = {};
  float mrow[8], lrow[8];
  for (int r = 0; r < 8; ++r) { mrow[r] = -1e30f; lrow[r] = 0.f; }

  const float scale = 0.125f;  // 1/sqrt(HD)
  const __bf16* khead = Kmat + (size_t)b * SEQ * HID + h * HD;  // +kv*HID
  const __bf16* vhead = Vt + ((size_t)b * NH + h) * HD * (size_t)SEQ;

#if USE_TDM
  tdm_load_2d((unsigned)(uintptr_t)&ktile[0][0][0], khead, HD, 32, HID);
  tdm_load_2d((unsigned)(uintptr_t)&vtile[0][0][0], vhead, 32, HD, SEQ);
#endif

  int cur = 0;
  for (int kv0 = 0; kv0 < SEQ; kv0 += 32, cur ^= 1) {
#if USE_TDM
    // WAR fence: previous-iteration LDS reads must land before the TDM
    // overwrites that buffer; then kick off the next tile pair.
    asm volatile("s_wait_dscnt 0" ::: "memory");
    if (kv0 + 32 < SEQ) {
      tdm_load_2d((unsigned)(uintptr_t)&ktile[cur ^ 1][0][0],
                  khead + (size_t)(kv0 + 32) * HID, HD, 32, HID);
      tdm_load_2d((unsigned)(uintptr_t)&vtile[cur ^ 1][0][0],
                  vhead + kv0 + 32, 32, HD, SEQ);
      __builtin_amdgcn_s_wait_tensorcnt(2);  // current pair done, next in flight
    } else {
      __builtin_amdgcn_s_wait_tensorcnt(0);
    }
#endif

    // ---- scores: S[16x32] = Q(16x64) @ K_tile^T, two 16x16 C fragments ----
    f32x8 s[2];
    for (int kt = 0; kt < 2; ++kt) {
#if USE_TDM
      const __bf16* krow = &ktile[cur][kt * 16 + lm][0];
#else
      const int kv = kv0 + kt * 16 + lm;
      const __bf16* krow = khead + (size_t)kv * HID;
      if (kv + 32 < SEQ) __builtin_prefetch(krow + (size_t)32 * HID, 0, 0);
#endif
      bf16x16 kb0 = ld_frag(krow + kbB,      krow + kbB + 16);
      bf16x16 kb1 = ld_frag(krow + 32 + kbB, krow + 32 + kbB + 16);
      f32x8 acc = {};
      acc = wmma_bf16(qa0, kb0, acc);
      acc = wmma_bf16(qa1, kb1, acc);
      s[kt] = acc;
    }

    // ---- scale + mask ----
    for (int kt = 0; kt < 2; ++kt) {
      const int kv = kv0 + kt * 16 + lm;
      const size_t mbase = ((size_t)b * SEQ + (q0 + half * 8)) * SEQ + kv;
      for (int r = 0; r < 8; ++r) {
        const int mv = mask[mbase + (size_t)r * SEQ];
        s[kt][r] = mv ? s[kt][r] * scale : -1e9f;
      }
    }

    // ---- online softmax (row = (vgpr r, wave half); reduce across 16 lanes)
    for (int r = 0; r < 8; ++r) {
      float v = fmaxf(s[0][r], s[1][r]);
      v = fmaxf(v, __shfl_xor(v, 1, 32));
      v = fmaxf(v, __shfl_xor(v, 2, 32));
      v = fmaxf(v, __shfl_xor(v, 4, 32));
      v = fmaxf(v, __shfl_xor(v, 8, 32));
      const float mnew  = fmaxf(mrow[r], v);
      const float alpha = __expf(mrow[r] - mnew);
      mrow[r] = mnew;
      const float p0 = __expf(s[0][r] - mnew);
      const float p1 = __expf(s[1][r] - mnew);
      s[0][r] = p0;
      s[1][r] = p1;
      float ps = p0 + p1;
      ps += __shfl_xor(ps, 1, 32);
      ps += __shfl_xor(ps, 2, 32);
      ps += __shfl_xor(ps, 4, 32);
      ps += __shfl_xor(ps, 8, 32);
      lrow[r] = lrow[r] * alpha + ps;
      for (int dt = 0; dt < 4; ++dt) oacc[dt][r] *= alpha;
    }

    // ---- re-shape P: C fragment -> LDS -> A fragment (16x32 bf16) ----
    for (int kt = 0; kt < 2; ++kt)
      for (int r = 0; r < 8; ++r)
        pbuf[half * 8 + r][kt * 16 + lm] = (__bf16)s[kt][r];
    asm volatile("s_wait_dscnt 0" ::: "memory");
    const bf16x16 pa = ld_frag(&pbuf[lm][kbA], &pbuf[lm][kbA + 16]);

    // ---- O(16x64) += P(16x32) @ V_tile(32x64) : 4 WMMAs over dim tiles ----
    for (int dt = 0; dt < 4; ++dt) {
#if USE_TDM
      const __bf16* vrow = &vtile[cur][dt * 16 + lm][kbB];
#else
      const __bf16* vrow = vhead + (size_t)(dt * 16 + lm) * SEQ + kv0 + kbB;
      if (kv0 + 32 < SEQ) __builtin_prefetch(vrow + 32, 0, 0);
#endif
      bf16x16 vb = ld_frag(vrow, vrow + 16);
      oacc[dt] = wmma_bf16(pa, vb, oacc[dt]);
    }
  }

  // ---- normalize + store ----
  for (int dt = 0; dt < 4; ++dt) {
    const int col = h * HD + dt * 16 + lm;
    for (int r = 0; r < 8; ++r) {
      const int q = q0 + half * 8 + r;
      const float inv = lrow[r] > 0.f ? 1.0f / lrow[r] : 0.f;
      O[(size_t)(b * SEQ + q) * HID + col] = (__bf16)(oacc[dt][r] * inv);
    }
  }
}

// ---------------------------------------------------------------------------
extern "C" void kernel_launch(void* const* d_in, const int* in_sizes, int n_in,
                              void* d_out, int out_size, void* d_ws,
                              size_t ws_size, hipStream_t stream) {
  const float* x    = (const float*)d_in[0];
  const int*   mask = (const int*)d_in[1];
  const float* Wq = (const float*)d_in[2];
  const float* bq = (const float*)d_in[3];
  const float* Wk = (const float*)d_in[4];
  const float* bk = (const float*)d_in[5];
  const float* Wv = (const float*)d_in[6];
  const float* bv = (const float*)d_in[7];
  const float* Wo = (const float*)d_in[8];
  const float* bo = (const float*)d_in[9];
  float* out = (float*)d_out;

  char* ws = (char*)d_ws;
  size_t off = 0;
  auto carve = [&](size_t bytes) -> void* {
    void* p = ws + off;
    off = (off + bytes + 255) & ~(size_t)255;
    return p;
  };

  const size_t MROWS = (size_t)BATCH * SEQ;  // 8192
  __bf16* Xb  = (__bf16*)carve(MROWS * HID * 2);
  __bf16* Wqb = (__bf16*)carve((size_t)HID * HID * 2);
  __bf16* Wkb = (__bf16*)carve((size_t)HID * HID * 2);
  __bf16* Wvb = (__bf16*)carve((size_t)HID * HID * 2);
  __bf16* Wob = (__bf16*)carve((size_t)HID * HID * 2);
  __bf16* Qb  = (__bf16*)carve(MROWS * HID * 2);
  __bf16* Kb  = (__bf16*)carve(MROWS * HID * 2);
  __bf16* Vt  = (__bf16*)carve(MROWS * HID * 2);  // [B, NH, HD, SEQ]
  __bf16* Ob  = (__bf16*)carve(MROWS * HID * 2);
  // total ~92 MB of d_ws

  const int nx = (int)(MROWS * HID);
  const int nw = HID * HID;
  cvt_f32_bf16_kernel<<<1024, 256, 0, stream>>>(x,  Xb,  nx);
  cvt_f32_bf16_kernel<<<512,  256, 0, stream>>>(Wq, Wqb, nw);
  cvt_f32_bf16_kernel<<<512,  256, 0, stream>>>(Wk, Wkb, nw);
  cvt_f32_bf16_kernel<<<512,  256, 0, stream>>>(Wv, Wvb, nw);
  cvt_f32_bf16_kernel<<<512,  256, 0, stream>>>(Wo, Wob, nw);

  dim3 gg((unsigned)(MROWS / 64), HID / 64);
  gemm_xwt_kernel<0><<<gg, 128, 0, stream>>>(Xb, Wqb, bq, Qb, (int)MROWS, HID, HID);
  gemm_xwt_kernel<0><<<gg, 128, 0, stream>>>(Xb, Wkb, bk, Kb, (int)MROWS, HID, HID);
  gemm_xwt_kernel<2><<<gg, 128, 0, stream>>>(Xb, Wvb, bv, Vt, (int)MROWS, HID, HID);

  dim3 ga(SEQ / 16, NH, BATCH);
  flash_attn_kernel<<<ga, 32, 0, stream>>>(Qb, Kb, Vt, mask, Ob);

  gemm_xwt_kernel<1><<<gg, 128, 0, stream>>>(Ob, Wob, bo, out, (int)MROWS, HID, HID);
}
